// RelationNet_30648886624750
// MI455X (gfx1250) — matrix operand order repaired and checked by reading, not compiled, not run on hardware
//
#include <hip/hip_runtime.h>
#include <hip/hip_bf16.h>

typedef __attribute__((ext_vector_type(16))) _Float16 v16h;
typedef __attribute__((ext_vector_type(8)))  float    v8f;

#define EPSV   1e-5f
#define SLOPE  0.2f

// Problem constants (from setup_inputs): B=2, N=2048, M=64, C=128, 2C=256, OUT=64
// Workspace layout (float offsets)
constexpr size_t OFF_HP     = 0;          // 2*2048*128
constexpr size_t OFF_HC     = 524288;     // 2*64*128
constexpr size_t OFF_A1     = 540672;     // 128
constexpr size_t OFF_C1     = 540800;     // 128
constexpr size_t OFF_SUM2   = 540928;     // 256 -- sum2..sumsq3 zeroed each launch
constexpr size_t OFF_SUMSQ2 = 541184;     // 256
constexpr size_t OFF_SUM3   = 541440;     // 256
constexpr size_t OFF_SUMSQ3 = 541696;     // 256
constexpr size_t OFF_A2     = 541952;     // 256
constexpr size_t OFF_C2     = 542208;     // 256
constexpr size_t OFF_A3     = 542464;     // 256
constexpr size_t OFF_C3     = 542720;     // 256
constexpr size_t OFF_A4     = 542976;     // 128
constexpr size_t OFF_C4     = 543104;     // 128
constexpr size_t OFF_A5     = 543232;     // 64
constexpr size_t OFF_C5     = 543296;     // 64
constexpr size_t OFF_POOLED = 543360;     // 2*2048*256
constexpr size_t OFF_Y1     = 1591936;    // 2*2048*128
constexpr size_t OFF_Y2     = 2116224;    // 2*2048*64
constexpr size_t OFF_W2P    = 2378368;    // 32768 halves (as 16384 floats)
constexpr size_t OFF_W3P    = 2394752;    // 65536 halves (as 32768 floats)

// ---------------------------------------------------------------------------
// hp[b,n,o] = sum_c point[b,n,c]*W1[o, c]      (o < 128, row length of W1 = 256)
// hc[b,m,o] = sum_c center[b,m,c]*W1[o, 128+c]
__global__ __launch_bounds__(128) void k_hphc(const float* __restrict__ pf,
                                              const float* __restrict__ cf,
                                              const float* __restrict__ W1,
                                              float* __restrict__ hp,
                                              float* __restrict__ hc) {
    __shared__ float xs[128];
    const int r = blockIdx.x;
    const float* x; float* out; int woff;
    if (r < 4096) { x = pf + (size_t)r * 128; out = hp + (size_t)r * 128; woff = 0; }
    else { int r2 = r - 4096; x = cf + (size_t)r2 * 128; out = hc + (size_t)r2 * 128; woff = 128; }
    xs[threadIdx.x] = x[threadIdx.x];
    __syncthreads();
    const int o = threadIdx.x;
    const float* wr = W1 + (size_t)o * 256 + woff;
    float acc = 0.f;
    #pragma unroll 8
    for (int c = 0; c < 128; ++c) acc += xs[c] * wr[c];
    out[o] = acc;
}

// ---------------------------------------------------------------------------
// Exact BN1 stats from hp/hc (separable decomposition of mean/var over (b,n,m))
__global__ __launch_bounds__(128) void k_bn1(const float* __restrict__ hp,
                                             const float* __restrict__ hc,
                                             const float* __restrict__ g1,
                                             const float* __restrict__ b1,
                                             float* __restrict__ aOut,
                                             float* __restrict__ cOut) {
    const int c = threadIdx.x;  // 128 channels
    float Shp = 0.f, Shp2 = 0.f, Shc = 0.f, Shc2 = 0.f, cross = 0.f;
    for (int b = 0; b < 2; ++b) {
        float sb = 0.f;
        for (int n = 0; n < 2048; ++n) {
            float v = hp[((size_t)(b * 2048 + n)) * 128 + c];
            sb += v; Shp2 += v * v;
        }
        float sc = 0.f;
        for (int m = 0; m < 64; ++m) {
            float v = hc[((size_t)(b * 64 + m)) * 128 + c];
            sc += v; Shc2 += v * v;
        }
        Shp += sb; Shc += sc; cross += sb * sc;
    }
    const float cnt  = 2.f * 2048.f * 64.f;
    const float mean = (64.f * Shp + 2048.f * Shc) / cnt;
    const float e2   = (64.f * Shp2 + 2048.f * Shc2 + 2.f * cross) / cnt;
    const float var  = e2 - mean * mean;
    const float a    = g1[c] * rsqrtf(var + EPSV);
    aOut[c] = a; cOut[c] = b1[c] - a * mean;
}

// ---------------------------------------------------------------------------
// Pack a (O x K) f32 row-major weight into f16 WMMA B-fragments:
// tile t = jt*(K/32)+kc covers K rows [kc*32, kc*32+32), N cols [jt*16, jt*16+16).
// Lane L holds column N = L&15, K = (h>>3)*16 + (L>>4)*8 + (h&7) (h = 0..15).
__global__ void k_pack(const float* __restrict__ W, _Float16* __restrict__ dst,
                       int O, int K) {
    const int total = O * K;
    for (int i = blockIdx.x * blockDim.x + threadIdx.x; i < total;
         i += gridDim.x * blockDim.x) {
        const int t = i >> 9, r = i & 511;
        const int lane = r >> 4, h = r & 15;
        const int KC = K >> 5;
        const int jt = t / KC, kc = t - jt * KC;
        const int o = jt * 16 + (lane & 15);
        const int k = kc * 32 + (h >> 3) * 16 + (lane >> 4) * 8 + (h & 7);
        dst[i] = (_Float16)W[(size_t)o * K + k];
    }
}

__global__ void k_zero(float* p, int n) {
    int i = blockIdx.x * blockDim.x + threadIdx.x;
    if (i < n) p[i] = 0.f;
}

// ---------------------------------------------------------------------------
// Fused layer1 -> layer2 -> layer3 pipeline, all matmuls via v_wmma_f32_16x16x32_f16.
// MODE 0: accumulate layer2 pre-act stats.  MODE 1: accumulate layer3 pre-act stats.
// MODE 2: full forward, max-pool over M -> pooled[b,n,256].
// Grid: 1024 blocks of 128 threads (4 waves); wave w handles n = (blk&511)*4+w, b = blk>>9.
// Templated on MODE (no runtime mode tests); kc3/mt loops kept rolled so VGPR
// use stays ~WMMA-accumulator-bound (=> ~4 waves/SIMD for cross-wave overlap).
template <int MODE>
__global__ __launch_bounds__(128) void k_fused(
    const float* __restrict__ hp, const float* __restrict__ hc,
    const float* __restrict__ a1, const float* __restrict__ c1,
    const _Float16* __restrict__ w2p, const _Float16* __restrict__ w3p,
    const float* __restrict__ bn2a, const float* __restrict__ bn2b,
    const float* __restrict__ bn3a, const float* __restrict__ bn3b,
    float* __restrict__ sumOut, float* __restrict__ sumsqOut,
    float* __restrict__ pooled) {
    __shared__ float    shc[64 * 132];        // padded stride 132 -> conflict-free
    __shared__ float    shp[4][128];
    __shared__ float    s2a[256], s2b[256], s3a[256], s3b[256];
    __shared__ _Float16 stage[4][512];        // per-wave 16x32 re-fragment buffer

    const int tid  = threadIdx.x;
    const int wave = tid >> 5;
    const int lane = tid & 31;
    const int l15  = lane & 15;
    const int hi   = lane >> 4;

    const int blk = blockIdx.x;
    const int b   = blk >> 9;
    const int n   = (blk & 511) * 4 + wave;

    // stage hc' = a1*hc + c1 (shared by all 4 waves), hp' = a1*hp (per wave)
    for (int i = tid; i < 64 * 128; i += 128) {
        const int m = i >> 7, ch = i & 127;
        shc[m * 132 + ch] = a1[ch] * hc[(size_t)(b * 64 + m) * 128 + ch] + c1[ch];
    }
    for (int ch = lane; ch < 128; ch += 32)
        shp[wave][ch] = a1[ch] * hp[(size_t)(b * 2048 + n) * 128 + ch];
    if (MODE >= 1)
        for (int i = tid; i < 256; i += 128) { s2a[i] = bn2a[i]; s2b[i] = bn2b[i]; }
    if (MODE == 2)
        for (int i = tid; i < 256; i += 128) { s3a[i] = bn3a[i]; s3b[i] = bn3b[i]; }
    __syncthreads();

    float statS[16], statQ[16], colmax[16];
    #pragma unroll
    for (int j = 0; j < 16; ++j) { statS[j] = 0.f; statQ[j] = 0.f; colmax[j] = -3.4e38f; }

    #pragma unroll 1
    for (int mt = 0; mt < 4; ++mt) {
        // Build layer-1 A fragments in registers (16-bit A layout: lane M=l15,
        // K(h) = 2*((h>>1)&3) + 16*(h>>3) + 8*hi + (h&1))
        v16h afrag[4];
        const int mrow = mt * 16 + l15;
        #pragma unroll
        for (int kc = 0; kc < 4; ++kc) {
            #pragma unroll
            for (int h = 0; h < 16; ++h) {
                const int k  = (((h >> 1) & 3) << 1) + ((h >> 3) << 4) + (hi << 3) + (h & 1);
                const int ch = kc * 32 + k;
                float v = shp[wave][ch] + shc[mrow * 132 + ch];
                v = v >= 0.f ? v : SLOPE * v;
                afrag[kc][h] = (_Float16)v;
            }
        }

        v8f acc3[16];
        if (MODE != 0) {
            v8f z = {};
            #pragma unroll
            for (int j = 0; j < 16; ++j) acc3[j] = z;
        }

        #pragma unroll 1
        for (int kc3 = 0; kc3 < 8; ++kc3) {
            if (kc3 < 7)  // warm next layer-3 weight tile (global_prefetch_b8)
                __builtin_prefetch(w3p + (size_t)(kc3 + 1) * 512 + lane * 16, 0, 1);

            #pragma unroll
            for (int jj = 0; jj < 2; ++jj) {
                const int j = kc3 * 2 + jj;            // layer-2 col-tile
                // two partial accumulators -> 2x WMMA ILP on the layer-2 chain
                v8f acc2a = {}, acc2b = {};
                {
                    v16h bf0 = *(const v16h*)(w2p + (size_t)(j * 4 + 0) * 512 + lane * 16);
                    v16h bf1 = *(const v16h*)(w2p + (size_t)(j * 4 + 1) * 512 + lane * 16);
                    v16h bf2 = *(const v16h*)(w2p + (size_t)(j * 4 + 2) * 512 + lane * 16);
                    v16h bf3 = *(const v16h*)(w2p + (size_t)(j * 4 + 3) * 512 + lane * 16);
                    acc2a = __builtin_amdgcn_wmma_f32_16x16x32_f16(
                        false, afrag[0], false, bf0, (short)0, acc2a, false, false);
                    acc2b = __builtin_amdgcn_wmma_f32_16x16x32_f16(
                        false, afrag[1], false, bf1, (short)0, acc2b, false, false);
                    acc2a = __builtin_amdgcn_wmma_f32_16x16x32_f16(
                        false, afrag[2], false, bf2, (short)0, acc2a, false, false);
                    acc2b = __builtin_amdgcn_wmma_f32_16x16x32_f16(
                        false, afrag[3], false, bf3, (short)0, acc2b, false, false);
                }
                v8f acc2 = acc2a + acc2b;

                if (MODE == 0) {
                    float s = 0.f, q = 0.f;
                    #pragma unroll
                    for (int r = 0; r < 8; ++r) { s += acc2[r]; q += acc2[r] * acc2[r]; }
                    statS[j] += s; statQ[j] += q;
                } else {
                    const int ch2 = j * 16 + l15;
                    const float ga = s2a[ch2], gb = s2b[ch2];
                    #pragma unroll
                    for (int r = 0; r < 8; ++r) {
                        float y = ga * acc2[r] + gb;
                        y = y >= 0.f ? y : SLOPE * y;
                        stage[wave][(r + 8 * hi) * 32 + jj * 16 + l15] = (_Float16)y;
                    }
                }
            }
            if (MODE != 0) {
                __syncthreads();  // order staging writes vs. re-fragment reads
                v16h a3frag;
                #pragma unroll
                for (int h = 0; h < 16; ++h) {
                    const int k = (((h >> 1) & 3) << 1) + ((h >> 3) << 4) + (hi << 3) + (h & 1);
                    a3frag[h] = stage[wave][l15 * 32 + k];
                }
                #pragma unroll
                for (int jt = 0; jt < 16; ++jt) {
                    v16h bfrag = *(const v16h*)(w3p + (size_t)(jt * 8 + kc3) * 512 + lane * 16);
                    acc3[jt] = __builtin_amdgcn_wmma_f32_16x16x32_f16(
                        false, a3frag, false, bfrag, (short)0, acc3[jt], false, false);
                }
            }
        }

        if (MODE == 1) {
            #pragma unroll
            for (int jt = 0; jt < 16; ++jt) {
                float s = 0.f, q = 0.f;
                #pragma unroll
                for (int r = 0; r < 8; ++r) { s += acc3[jt][r]; q += acc3[jt][r] * acc3[jt][r]; }
                statS[jt] += s; statQ[jt] += q;
            }
        } else if (MODE == 2) {
            #pragma unroll
            for (int jt = 0; jt < 16; ++jt) {
                const int ch3 = jt * 16 + l15;
                const float ga = s3a[ch3], gb = s3b[ch3];
                float mm = -3.4e38f;
                #pragma unroll
                for (int r = 0; r < 8; ++r) {
                    float y = ga * acc3[jt][r] + gb;
                    y = y >= 0.f ? y : SLOPE * y;
                    mm = fmaxf(mm, y);
                }
                colmax[jt] = fmaxf(colmax[jt], mm);
            }
        }
    }

    if (MODE == 2) {
        #pragma unroll
        for (int jt = 0; jt < 16; ++jt) {
            float v = fmaxf(colmax[jt], __shfl_xor(colmax[jt], 16, 32));
            if (lane < 16)
                pooled[(size_t)(b * 2048 + n) * 256 + jt * 16 + lane] = v;
        }
    } else {
        #pragma unroll
        for (int jt = 0; jt < 16; ++jt) {
            float s = statS[jt] + __shfl_xor(statS[jt], 16, 32);
            float q = statQ[jt] + __shfl_xor(statQ[jt], 16, 32);
            if (lane < 16) {
                atomicAdd(&sumOut[jt * 16 + lane], s);
                atomicAdd(&sumsqOut[jt * 16 + lane], q);
            }
        }
    }
}

// ---------------------------------------------------------------------------
__global__ void k_finalize(const float* __restrict__ sum, const float* __restrict__ sumsq,
                           const float* __restrict__ g, const float* __restrict__ beta,
                           float cnt, int nch, float* __restrict__ aOut, float* __restrict__ cOut) {
    const int c = blockIdx.x * blockDim.x + threadIdx.x;
    if (c < nch) {
        const float mean = sum[c] / cnt;
        const float var  = sumsq[c] / cnt - mean * mean;
        const float a    = g[c] * rsqrtf(var + EPSV);
        aOut[c] = a; cOut[c] = beta[c] - a * mean;
    }
}

// Per-channel stats + BN coefficients for the FC head (one block per channel)
__global__ __launch_bounds__(256) void k_channel_bn(const float* __restrict__ x,
                                                    int rows, int nch,
                                                    const float* __restrict__ g,
                                                    const float* __restrict__ beta,
                                                    float* __restrict__ aOut,
                                                    float* __restrict__ cOut) {
    __shared__ float ss[256], sq[256];
    const int c = blockIdx.x;
    float s = 0.f, q = 0.f;
    for (int r = threadIdx.x; r < rows; r += 256) {
        const float v = x[(size_t)r * nch + c];
        s += v; q += v * v;
    }
    ss[threadIdx.x] = s; sq[threadIdx.x] = q;
    __syncthreads();
    for (int off = 128; off > 0; off >>= 1) {
        if (threadIdx.x < off) {
            ss[threadIdx.x] += ss[threadIdx.x + off];
            sq[threadIdx.x] += sq[threadIdx.x + off];
        }
        __syncthreads();
    }
    if (threadIdx.x == 0) {
        const float mean = ss[0] / rows;
        const float var  = sq[0] / rows - mean * mean;
        const float a    = g[c] * rsqrtf(var + EPSV);
        aOut[c] = a; cOut[c] = beta[c] - a * mean;
    }
}

__global__ __launch_bounds__(128) void k_fc1(const float* __restrict__ pooled,
                                             const float* __restrict__ w,
                                             const float* __restrict__ bias,
                                             float* __restrict__ y) {
    __shared__ float xs[256];
    const int row = blockIdx.x;
    for (int i = threadIdx.x; i < 256; i += 128) xs[i] = pooled[(size_t)row * 256 + i];
    __syncthreads();
    const int o = threadIdx.x;
    const float* wr = w + (size_t)o * 256;
    float acc = bias[o];
    #pragma unroll 8
    for (int k = 0; k < 256; ++k) acc += xs[k] * wr[k];
    y[(size_t)row * 128 + o] = acc;
}

__global__ __launch_bounds__(128) void k_fc2(const float* __restrict__ y1,
                                             const float* __restrict__ a4,
                                             const float* __restrict__ c4,
                                             const float* __restrict__ w,
                                             const float* __restrict__ bias,
                                             float* __restrict__ y2) {
    __shared__ float xs[128];
    const int row = blockIdx.x;
    const int t = threadIdx.x;
    { float v = a4[t] * y1[(size_t)row * 128 + t] + c4[t]; xs[t] = v > 0.f ? v : 0.f; }
    __syncthreads();
    if (t < 64) {
        const float* wr = w + (size_t)t * 128;
        float acc = bias[t];
        #pragma unroll 8
        for (int k = 0; k < 128; ++k) acc += xs[k] * wr[k];
        y2[(size_t)row * 64 + t] = acc;
    }
}

__global__ void k_out(const float* __restrict__ y2, const float* __restrict__ a5,
                      const float* __restrict__ c5, float* __restrict__ out, int total) {
    const int i = blockIdx.x * blockDim.x + threadIdx.x;
    if (i < total) {
        const int o = i & 63;
        const float v = a5[o] * y2[i] + c5[o];
        out[i] = v > 0.f ? v : 0.f;
    }
}

// ---------------------------------------------------------------------------
extern "C" void kernel_launch(void* const* d_in, const int* in_sizes, int n_in,
                              void* d_out, int out_size, void* d_ws, size_t ws_size,
                              hipStream_t stream) {
    const float* pf   = (const float*)d_in[0];
    const float* cf   = (const float*)d_in[1];
    const float* W1   = (const float*)d_in[2];
    const float* g1   = (const float*)d_in[3];
    const float* b1   = (const float*)d_in[4];
    const float* W2   = (const float*)d_in[5];
    const float* g2   = (const float*)d_in[6];
    const float* b2   = (const float*)d_in[7];
    const float* W3   = (const float*)d_in[8];
    const float* g3   = (const float*)d_in[9];
    const float* b3   = (const float*)d_in[10];
    const float* fc1w = (const float*)d_in[11];
    const float* fc1b = (const float*)d_in[12];
    const float* g4   = (const float*)d_in[13];
    const float* b4   = (const float*)d_in[14];
    const float* fc2w = (const float*)d_in[15];
    const float* fc2b = (const float*)d_in[16];
    const float* g5   = (const float*)d_in[17];
    const float* b5   = (const float*)d_in[18];

    float* ws = (float*)d_ws;
    float* hp   = ws + OFF_HP;   float* hc    = ws + OFF_HC;
    float* a1   = ws + OFF_A1;   float* c1    = ws + OFF_C1;
    float* sum2 = ws + OFF_SUM2; float* ssq2  = ws + OFF_SUMSQ2;
    float* sum3 = ws + OFF_SUM3; float* ssq3  = ws + OFF_SUMSQ3;
    float* a2   = ws + OFF_A2;   float* c2    = ws + OFF_C2;
    float* a3   = ws + OFF_A3;   float* c3    = ws + OFF_C3;
    float* a4   = ws + OFF_A4;   float* c4    = ws + OFF_C4;
    float* a5   = ws + OFF_A5;   float* c5    = ws + OFF_C5;
    float* pooled = ws + OFF_POOLED;
    float* y1   = ws + OFF_Y1;   float* y2    = ws + OFF_Y2;
    _Float16* w2p = (_Float16*)(ws + OFF_W2P);
    _Float16* w3p = (_Float16*)(ws + OFF_W3P);

    // Stage A: projections + exact BN1 stats + weight packing
    k_hphc<<<4224, 128, 0, stream>>>(pf, cf, W1, hp, hc);
    k_bn1<<<1, 128, 0, stream>>>(hp, hc, g1, b1, a1, c1);
    k_pack<<<128, 256, 0, stream>>>(W2, w2p, 256, 128);
    k_pack<<<256, 256, 0, stream>>>(W3, w3p, 256, 256);
    k_zero<<<4, 256, 0, stream>>>(sum2, 1024);  // sum2/ssq2/sum3/ssq3 contiguous

    // Stage B: fused WMMA pipeline, 3 passes (stats2, stats3, forward+maxpool)
    k_fused<0><<<1024, 128, 0, stream>>>(hp, hc, a1, c1, w2p, w3p, a2, c2, a3, c3,
                                         sum2, ssq2, pooled);
    k_finalize<<<1, 256, 0, stream>>>(sum2, ssq2, g2, b2, 262144.f, 256, a2, c2);
    k_fused<1><<<1024, 128, 0, stream>>>(hp, hc, a1, c1, w2p, w3p, a2, c2, a3, c3,
                                         sum3, ssq3, pooled);
    k_finalize<<<1, 256, 0, stream>>>(sum3, ssq3, g3, b3, 262144.f, 256, a3, c3);
    k_fused<2><<<1024, 128, 0, stream>>>(hp, hc, a1, c1, w2p, w3p, a2, c2, a3, c3,
                                         sum3, ssq3, pooled);

    // Stage C: FC head with BN over (b,n)
    k_fc1<<<4096, 128, 0, stream>>>(pooled, fc1w, fc1b, y1);
    k_channel_bn<<<128, 256, 0, stream>>>(y1, 4096, 128, g4, b4, a4, c4);
    k_fc2<<<4096, 128, 0, stream>>>(y1, a4, c4, fc2w, fc2b, y2);
    k_channel_bn<<<64, 256, 0, stream>>>(y2, 4096, 64, g5, b5, a5, c5);
    k_out<<<1024, 256, 0, stream>>>(y2, a5, c5, (float*)d_out, 262144);
}
